// SparseBEVTransformerDecoder_12446815224512
// MI455X (gfx1250) — compile-verified, matrix operand order checked by reading, not run on hardware
//
#include <hip/hip_runtime.h>

// ---------------------------------------------------------------------------
// SparseBEV decoder for MI455X (gfx1250).
// - bf16 WMMA (v_wmma_f32_16x16x32_bf16) everywhere, fp32 accumulation
// - B matrices pre-packed into per-lane WMMA fragment order (32B/lane/kstep)
// - A tiles double-buffered in LDS via global_load_async_to_lds_b128 with
//   s_wait_asynccnt 0x2 prefetch pipelining (copy chunk i+1 under compute i)
// - feature pyramids repacked (BG,N,H,W,C) bf16 so bilinear corners are
//   contiguous 128B reads across the 64-channel block
// - LDS reductions for layernorm
// ---------------------------------------------------------------------------

typedef __bf16 bf16_t;
typedef __attribute__((ext_vector_type(16))) __bf16 v16bf;
typedef __attribute__((ext_vector_type(8)))  __bf16 v8bf;
typedef __attribute__((ext_vector_type(8)))  float  v8f;

#define BATCH 2
#define QN    900
#define GG    4
#define PP    8
#define OPP   32
#define CEE   64
#define NVIEW 6
#define BQ    (BATCH*QN)       // 1800
#define TOT   4352             // CE*CE + P*OP
#define NPAR  (GG*TOT)         // 17408
#define NPTS  (BATCH*QN*GG*PP) // 57600
#define KCHUNK 256             // LDS A-tile depth (bf16): 16 x 256 x 2 bufs = 16KB

// ---------------- elementwise f32 -> bf16 ----------------
__global__ void f32_to_bf16_kernel(const float* __restrict__ in, bf16_t* __restrict__ out, int n) {
  int i = blockIdx.x * blockDim.x + threadIdx.x;
  if (i < n) out[i] = (bf16_t)in[i];
}

// ---------------- pack row-major f32 B(KxN) into bf16 WMMA-fragment order ----
// PB[(((nt*Kt + kt)*32 + lane)*16 + e]; e maps to the ISA 16-bit 16x32 layout:
// lanes<16 carry K {0..7,16..23}, lanes>=16 carry K {8..15,24..31}.
__global__ void pack_b_kernel(const float* __restrict__ Bsrc, bf16_t* __restrict__ PB,
                              int Kdim, int Ndim) {
  int i = blockIdx.x * blockDim.x + threadIdx.x;
  if (i >= Kdim * Ndim) return;
  int e    = i & 15;
  int lane = (i >> 4) & 31;
  int t    = i >> 9;
  int Kt   = Kdim >> 5;
  int kt   = t % Kt;
  int nt   = t / Kt;
  int half = lane >> 4, mr = lane & 15;
  int k = kt * 32 + ((e < 8) ? (half * 8 + e) : (16 + half * 8 + (e - 8)));
  int n = nt * 16 + mr;
  PB[i] = (bf16_t)Bsrc[(size_t)k * Ndim + n];
}

// ---------------- repack feature level (BG,C,N,H,W) f32 -> (BG,N,H,W,C) bf16 --
__global__ void repack_feat_kernel(const float* __restrict__ in, bf16_t* __restrict__ outp,
                                   int H, int W) {
  int o = blockIdx.x * blockDim.x + threadIdx.x;
  int total = BATCH * GG * NVIEW * H * W * CEE;
  if (o >= total) return;
  int c = o & 63;
  int t = o >> 6;
  int w = t % W; t /= W;
  int h = t % H; t /= H;
  int n = t % NVIEW; t /= NVIEW;
  int bg = t;
  outp[o] = (bf16_t)in[((((size_t)bg * CEE + c) * NVIEW + n) * H + h) * W + w];
}

// ---------------- projection + view-select + scale softmax ----------------
__global__ void project_softmax_kernel(const float* __restrict__ pts,
                                       const float* __restrict__ l2i,
                                       const float* __restrict__ scw,
                                       float* __restrict__ ptsbuf,
                                       float* __restrict__ swbuf) {
  int tid = blockIdx.x * blockDim.x + threadIdx.x;
  if (tid >= NPTS) return;
  int p = tid & 7;
  int g = (tid >> 3) & 3;
  int q = (tid >> 5) % QN;
  int b = tid / (QN * 32);
  const float* pt = pts + (size_t)tid * 3;
  float X = pt[0], Y = pt[1], Z = pt[2];
  float selx = 0.f, sely = 0.f;
  int   iv = 0;
  bool  found = false;
  for (int n = 0; n < NVIEW; n++) {
    const float* L = l2i + (size_t)(b * NVIEW + n) * 16;
    float c0 = L[0]*X + L[1]*Y + L[2]*Z  + L[3];
    float c1 = L[4]*X + L[5]*Y + L[6]*Z  + L[7];
    float c2 = L[8]*X + L[9]*Y + L[10]*Z + L[11];
    float h  = fmaxf(c2, 1e-5f);
    float xn = (c0 / h) / 1408.0f;
    float yn = (c1 / h) / 512.0f;
    bool valid = (c2 > 1e-5f) && (xn > 0.f) && (xn < 1.f) && (yn > 0.f) && (yn < 1.f);
    if (n == 0) { selx = xn; sely = yn; }
    if (valid && !found) { iv = n; selx = xn; sely = yn; found = true; }
  }
  int oidx = ((b * GG + g) * QN + q) * PP + p;
  ptsbuf[oidx*4+0] = selx;
  ptsbuf[oidx*4+1] = sely;
  ptsbuf[oidx*4+2] = (float)iv;
  const float* w = scw + (size_t)tid * 4;
  float m  = fmaxf(fmaxf(w[0], w[1]), fmaxf(w[2], w[3]));
  float e0 = expf(w[0]-m), e1 = expf(w[1]-m), e2 = expf(w[2]-m), e3 = expf(w[3]-m);
  float s  = e0 + e1 + e2 + e3;
  swbuf[oidx*4+0] = e0/s; swbuf[oidx*4+1] = e1/s;
  swbuf[oidx*4+2] = e2/s; swbuf[oidx*4+3] = e3/s;
}

// ---------------- multi-level bilinear sampling (packed bf16 feats) ----------
__global__ void sample_kernel(const bf16_t* __restrict__ f0, const bf16_t* __restrict__ f1,
                              const bf16_t* __restrict__ f2, const bf16_t* __restrict__ f3,
                              const float* __restrict__ ptsbuf, const float* __restrict__ swbuf,
                              bf16_t* __restrict__ xout) {
  int idx = blockIdx.x;          // ((bg)*QN + q)*PP + p
  int c   = threadIdx.x;         // channel 0..63
  int p   = idx & 7;
  int q   = (idx >> 3) % QN;
  int bg  = idx / (QN * PP);
  float sx = ptsbuf[idx*4], sy = ptsbuf[idx*4+1];
  int   iv = (int)ptsbuf[idx*4+2];
  const bf16_t* feats[4] = { f0, f1, f2, f3 };
  const int HH[4] = {64, 32, 16, 8};
  const int WW[4] = {176, 88, 44, 22};
  float acc = 0.f;
  #pragma unroll
  for (int l = 0; l < 4; l++) {
    int H = HH[l], W = WW[l];
    float xx = sx * W - 0.5f, yy = sy * H - 0.5f;
    float x0f = floorf(xx), y0f = floorf(yy);
    float wx = xx - x0f, wy = yy - y0f;
    int x0 = (int)x0f, y0 = (int)y0f;
    int x0i = min(max(x0,     0), W-1), x1i = min(max(x0 + 1, 0), W-1);
    int y0i = min(max(y0,     0), H-1), y1i = min(max(y0 + 1, 0), H-1);
    // layout (BG,N,H,W,C): 64-channel block contiguous -> 128B coalesced reads
    const bf16_t* fb = feats[l] + (((size_t)bg * NVIEW + iv) * (size_t)(H * W)) * CEE + c;
    float v00 = (float)fb[((size_t)y0i * W + x0i) * CEE];
    float v01 = (float)fb[((size_t)y0i * W + x1i) * CEE];
    float v10 = (float)fb[((size_t)y1i * W + x0i) * CEE];
    float v11 = (float)fb[((size_t)y1i * W + x1i) * CEE];
    float bil = (1.f-wy) * ((1.f-wx)*v00 + wx*v01) + wy * ((1.f-wx)*v10 + wx*v11);
    acc += swbuf[idx*4 + l] * bil;
  }
  int b = bg / GG, g = bg % GG, bq = b * QN + q;
  xout[((size_t)(bq * GG + g) * PP + p) * CEE + c] = (bf16_t)acc;
}

// ---------------- generic WMMA GEMM: packed B, double-buffered async-LDS A ---
// Block = 256 threads (8 waves). One m-tile per block, 16 n-tiles (2 per wave).
// A chunk (16 x KCHUNK bf16) prefetched for chunk i+1 while computing chunk i.
__global__ void __launch_bounds__(256)
wmma_gemm_kernel(const bf16_t* __restrict__ A, const bf16_t* __restrict__ PB,
                 const float* __restrict__ bias, const float* __restrict__ resid,
                 float* __restrict__ outF, bf16_t* __restrict__ outB,
                 int Mdim, int Ndim, int Kdim) {
  __shared__ __align__(32) bf16_t Atile[2][16 * KCHUNK];   // ping-pong, 16KB
  int tid  = threadIdx.x;
  int lane = tid & 31, wave = tid >> 5;
  int half = lane >> 4, mr = lane & 15;
  int mt   = blockIdx.y;
  int nt0  = (blockIdx.x * 8 + wave) * 2;     // two n-tiles per wave
  int Kt   = Kdim >> 5;
  int ko1  = half * 8, ko2 = 16 + half * 8;
  int sr   = tid >> 4;                        // staged row 0..15
  int sc   = (tid & 15) * 16;                 // 32B segment (16 elems)
  int grow = mt * 16 + sr;
  int nchunks = Kdim / KCHUNK;

  auto stage = [&](int ci, int buf) {
    bf16_t* ldst = &Atile[buf][sr * KCHUNK + sc];
    if (grow < Mdim) {
      unsigned loff = (unsigned)(unsigned long long)(uintptr_t)ldst;
      unsigned long long gp =
          (unsigned long long)(uintptr_t)(A + (size_t)grow * Kdim + ci * KCHUNK + sc);
      // INST_OFFSET is added to both LDS and global addresses (CDNA5 ISA 10.2)
      asm volatile("global_load_async_to_lds_b128 %0, %1, off\n\t"
                   "global_load_async_to_lds_b128 %0, %1, off offset:16"
                   :: "v"(loff), "v"(gp) : "memory");
    } else {
      v8bf z;
      #pragma unroll
      for (int j = 0; j < 8; j++) z[j] = (bf16_t)0.f;
      *(v8bf*)ldst       = z;
      *(v8bf*)(ldst + 8) = z;
    }
  };

  v8f acc0 = {0.f,0.f,0.f,0.f,0.f,0.f,0.f,0.f};
  v8f acc1 = {0.f,0.f,0.f,0.f,0.f,0.f,0.f,0.f};
  stage(0, 0);
  for (int ci = 0; ci < nchunks; ci++) {
    int buf = ci & 1;
    if (ci + 1 < nchunks) {
      __syncthreads();                 // readers done with Atile[buf^1] (chunk ci-1)
      stage(ci + 1, buf ^ 1);          // prefetch next chunk under compute
      // async loads retire in order: <=2 outstanding => chunk ci has landed
      asm volatile("s_wait_asynccnt 0x2" ::: "memory");
    } else {
      asm volatile("s_wait_asynccnt 0x0" ::: "memory");
    }
    __syncthreads();                   // publish chunk ci
    const bf16_t* Ab = Atile[buf];
    #pragma unroll
    for (int ks = 0; ks < KCHUNK / 32; ks++) {
      v8bf lo = *(const v8bf*)&Ab[mr * KCHUNK + ks * 32 + ko1];
      v8bf hi = *(const v8bf*)&Ab[mr * KCHUNK + ks * 32 + ko2];
      v16bf a;
      #pragma unroll
      for (int j = 0; j < 8; j++) { a[j] = lo[j]; a[8 + j] = hi[j]; }
      int ktg = (ci * (KCHUNK / 32)) + ks;
      v16bf b0 = *(const v16bf*)(PB + (((size_t)nt0       * Kt + ktg) * 32 + lane) * 16);
      v16bf b1 = *(const v16bf*)(PB + (((size_t)(nt0 + 1) * Kt + ktg) * 32 + lane) * 16);
      acc0 = __builtin_amdgcn_wmma_f32_16x16x32_bf16(false, a, false, b0, (short)0, acc0, false, false);
      acc1 = __builtin_amdgcn_wmma_f32_16x16x32_bf16(false, a, false, b1, (short)0, acc1, false, false);
    }
  }
  #pragma unroll
  for (int r = 0; r < 8; r++) {
    int rowD = mt * 16 + r + half * 8;
    if (rowD < Mdim) {
      int col0 = nt0 * 16 + mr, col1 = col0 + 16;
      float v0 = acc0[r], v1 = acc1[r];
      if (bias)  { v0 += bias[col0];  v1 += bias[col1]; }
      if (resid) { v0 += resid[(size_t)rowD * Ndim + col0]; v1 += resid[(size_t)rowD * Ndim + col1]; }
      if (outF) { outF[(size_t)rowD * Ndim + col0] = v0; outF[(size_t)rowD * Ndim + col1] = v1; }
      else      { outB[(size_t)rowD * Ndim + col0] = (bf16_t)v0; outB[(size_t)rowD * Ndim + col1] = (bf16_t)v1; }
    }
  }
}

// ---------------- per-(bq,g): out1(8x64) = x(8x64) @ M(64x64) ----------------
// 4 waves/block share M staged transposed in LDS -> contiguous 16B B-frag reads.
__global__ void __launch_bounds__(128)
xm_gemm_kernel(const bf16_t* __restrict__ xin, const bf16_t* __restrict__ params,
               float* __restrict__ out1) {
  __shared__ __align__(32) bf16_t Mt[64 * 64];  // [n][k]
  int tid = threadIdx.x;
  int lane = tid & 31, wave = tid >> 5;   // wave == n-tile 0..3
  int half = lane >> 4, mr = lane & 15;
  int bqg = blockIdx.x;
  int bq = bqg >> 2, g = bqg & 3;
  const bf16_t* Mg = params + (size_t)bq * NPAR + g * TOT;   // 64x64 row-major [k][n]
  for (int i = tid * 8; i < 4096; i += 128 * 8) {            // vectorized transpose
    int k = i >> 6, n = i & 63;
    v8bf vv = *(const v8bf*)&Mg[i];
    #pragma unroll
    for (int j = 0; j < 8; j++) Mt[(n + j) * 64 + k] = vv[j];
  }
  __syncthreads();
  const bf16_t* A = xin + (size_t)bqg * 512;                 // 8x64
  int col = wave * 16 + mr;
  int ko1 = half * 8, ko2 = 16 + half * 8;
  v8f acc = {0.f,0.f,0.f,0.f,0.f,0.f,0.f,0.f};
  #pragma unroll
  for (int kt = 0; kt < 2; kt++) {
    int k0 = kt * 32;
    v16bf a, b;
    if (mr < 8) {
      v8bf alo = *(const v8bf*)&A[mr * 64 + k0 + ko1];
      v8bf ahi = *(const v8bf*)&A[mr * 64 + k0 + ko2];
      #pragma unroll
      for (int j = 0; j < 8; j++) { a[j] = alo[j]; a[8 + j] = ahi[j]; }
    } else {
      #pragma unroll
      for (int j = 0; j < 16; j++) a[j] = (bf16_t)0.f;
    }
    v8bf blo = *(const v8bf*)&Mt[col * 64 + k0 + ko1];
    v8bf bhi = *(const v8bf*)&Mt[col * 64 + k0 + ko2];
    #pragma unroll
    for (int j = 0; j < 8; j++) { b[j] = blo[j]; b[8 + j] = bhi[j]; }
    acc = __builtin_amdgcn_wmma_f32_16x16x32_bf16(false, a, false, b, (short)0, acc, false, false);
  }
  if (half == 0) {
    #pragma unroll
    for (int r = 0; r < 8; r++) out1[(size_t)bqg * 512 + r * 64 + col] = acc[r];
  }
}

// ---------------- per-(bq,g): out2(32x64) = S(32x8) @ y(8x64), K pad 8->32 ----
// y is consumed transposed (y1t[ce][p]) so both fragments are 16B contiguous.
__global__ void sy_gemm_kernel(const bf16_t* __restrict__ params, const bf16_t* __restrict__ y1t,
                               float* __restrict__ out2) {
  int lane = threadIdx.x, half = lane >> 4, mr = lane & 15;
  int t = blockIdx.x;           // 0..7 : mt(2) x nt(4)
  int mt = t >> 2, nt = t & 3;
  int bqg = blockIdx.y;
  int bq = bqg >> 2, g = bqg & 3;
  const bf16_t* S = params + (size_t)bq * NPAR + g * TOT + 4096;  // 32x8 row-major
  const bf16_t* Y = y1t + (size_t)bqg * 512;                      // [ce][p] 64x8
  int col = nt * 16 + mr;
  int row = mt * 16 + mr;
  v16bf a, b;
  #pragma unroll
  for (int j = 0; j < 16; j++) { a[j] = (bf16_t)0.f; b[j] = (bf16_t)0.f; }
  if (half == 0) {  // lanes<16 carry K=0..7 in elems 0..7; rest stays zero
    v8bf sa = *(const v8bf*)&S[row * 8];
    v8bf yb = *(const v8bf*)&Y[col * 8];
    #pragma unroll
    for (int j = 0; j < 8; j++) { a[j] = sa[j]; b[j] = yb[j]; }
  }
  v8f acc = {0.f,0.f,0.f,0.f,0.f,0.f,0.f,0.f};
  acc = __builtin_amdgcn_wmma_f32_16x16x32_bf16(false, a, false, b, (short)0, acc, false, false);
  #pragma unroll
  for (int r = 0; r < 8; r++) {
    int rowD = mt * 16 + r + half * 8;
    out2[(size_t)bqg * 2048 + rowD * 64 + col] = acc[r];
  }
}

// ---------------- layernorm (+ReLU) over nvals per block -> bf16 -------------
__global__ void ln_relu_kernel(const float* __restrict__ in, bf16_t* __restrict__ out, int nvals) {
  __shared__ float red[256];
  int gidx = blockIdx.x;
  const float* base = in + (size_t)gidx * nvals;
  float s = 0.f;
  for (int i = threadIdx.x; i < nvals; i += blockDim.x) s += base[i];
  red[threadIdx.x] = s; __syncthreads();
  for (int st = 128; st > 0; st >>= 1) {
    if ((int)threadIdx.x < st) red[threadIdx.x] += red[threadIdx.x + st];
    __syncthreads();
  }
  float mean = red[0] / (float)nvals;
  __syncthreads();
  float sq = 0.f;
  for (int i = threadIdx.x; i < nvals; i += blockDim.x) { float d = base[i] - mean; sq += d * d; }
  red[threadIdx.x] = sq; __syncthreads();
  for (int st = 128; st > 0; st >>= 1) {
    if ((int)threadIdx.x < st) red[threadIdx.x] += red[threadIdx.x + st];
    __syncthreads();
  }
  float rstd = rsqrtf(red[0] / (float)nvals + 1e-5f);
  for (int i = threadIdx.x; i < nvals; i += blockDim.x) {
    float v = (base[i] - mean) * rstd;
    out[(size_t)gidx * nvals + i] = (bf16_t)fmaxf(v, 0.f);
  }
}

// LN1 variant: 512 vals = p(8) x ce(64); writes y transposed as [ce][p].
__global__ void ln_relu_xpose_kernel(const float* __restrict__ in, bf16_t* __restrict__ outT) {
  __shared__ float red[256];
  int gidx = blockIdx.x;
  const float* base = in + (size_t)gidx * 512;
  float s = 0.f;
  for (int i = threadIdx.x; i < 512; i += blockDim.x) s += base[i];
  red[threadIdx.x] = s; __syncthreads();
  for (int st = 128; st > 0; st >>= 1) {
    if ((int)threadIdx.x < st) red[threadIdx.x] += red[threadIdx.x + st];
    __syncthreads();
  }
  float mean = red[0] / 512.f;
  __syncthreads();
  float sq = 0.f;
  for (int i = threadIdx.x; i < 512; i += blockDim.x) { float d = base[i] - mean; sq += d * d; }
  red[threadIdx.x] = sq; __syncthreads();
  for (int st = 128; st > 0; st >>= 1) {
    if ((int)threadIdx.x < st) red[threadIdx.x] += red[threadIdx.x + st];
    __syncthreads();
  }
  float rstd = rsqrtf(red[0] / 512.f + 1e-5f);
  for (int i = threadIdx.x; i < 512; i += blockDim.x) {
    float v = (base[i] - mean) * rstd;
    int p = i >> 6, ce = i & 63;
    outT[(size_t)gidx * 512 + ce * 8 + p] = (bf16_t)fmaxf(v, 0.f);
  }
}

// ---------------------------------------------------------------------------
extern "C" void kernel_launch(void* const* d_in, const int* in_sizes, int n_in,
                              void* d_out, int out_size, void* d_ws, size_t ws_size,
                              hipStream_t stream) {
  const float* query = (const float*)d_in[0];
  const float* spts  = (const float*)d_in[1];
  const float* scw   = (const float*)d_in[2];
  const float* l2i   = (const float*)d_in[3];
  const float* f0    = (const float*)d_in[4];
  const float* f1    = (const float*)d_in[5];
  const float* f2    = (const float*)d_in[6];
  const float* f3    = (const float*)d_in[7];
  const float* pg_w  = (const float*)d_in[8];
  const float* pg_b  = (const float*)d_in[9];
  const float* out_w = (const float*)d_in[10];
  const float* out_b = (const float*)d_in[11];
  float* out = (float*)d_out;

  char* ws = (char*)d_ws;
  size_t off = 0;
  auto alloc = [&](size_t bytes) -> void* {
    void* p = ws + off;
    off = (off + bytes + 255) & ~(size_t)255;
    return p;
  };
  const size_t HW0 = 64*176, HW1 = 32*88, HW2 = 16*44, HW3 = 8*22;
  bf16_t* q_bf    = (bf16_t*)alloc((size_t)BQ * 256 * 2);
  bf16_t* pgw_pk  = (bf16_t*)alloc((size_t)256 * NPAR * 2);   // packed fragments
  bf16_t* outw_pk = (bf16_t*)alloc((size_t)8192 * 256 * 2);   // packed fragments
  bf16_t* params  = (bf16_t*)alloc((size_t)BQ * NPAR * 2);
  float*  ptsbuf  = (float*) alloc((size_t)NPTS * 4 * 4);
  float*  swbuf   = (float*) alloc((size_t)NPTS * 4 * 4);
  bf16_t* x_bf    = (bf16_t*)alloc((size_t)BQ * GG * PP * CEE * 2);
  float*  out1    = (float*) alloc((size_t)BQ * GG * 512 * 4);
  bf16_t* y1t     = (bf16_t*)alloc((size_t)BQ * GG * 512 * 2);
  float*  out2    = (float*) alloc((size_t)BQ * GG * 2048 * 4);
  bf16_t* xf      = (bf16_t*)alloc((size_t)BQ * 8192 * 2);
  bf16_t* fp0     = (bf16_t*)alloc((size_t)8 * NVIEW * HW0 * CEE * 2);
  bf16_t* fp1     = (bf16_t*)alloc((size_t)8 * NVIEW * HW1 * CEE * 2);
  bf16_t* fp2     = (bf16_t*)alloc((size_t)8 * NVIEW * HW2 * CEE * 2);
  bf16_t* fp3     = (bf16_t*)alloc((size_t)8 * NVIEW * HW3 * CEE * 2);

  // precision conversion / fragment packing / feature repack
  f32_to_bf16_kernel<<<(BQ*256 + 255)/256, 256, 0, stream>>>(query, q_bf, BQ*256);
  pack_b_kernel<<<(256*NPAR + 255)/256, 256, 0, stream>>>(pg_w, pgw_pk, 256, NPAR);
  pack_b_kernel<<<(8192*256 + 255)/256, 256, 0, stream>>>(out_w, outw_pk, 8192, 256);
  {
    int t0 = 8*NVIEW*(int)HW0*CEE, t1 = 8*NVIEW*(int)HW1*CEE;
    int t2 = 8*NVIEW*(int)HW2*CEE, t3 = 8*NVIEW*(int)HW3*CEE;
    repack_feat_kernel<<<(t0 + 255)/256, 256, 0, stream>>>(f0, fp0, 64, 176);
    repack_feat_kernel<<<(t1 + 255)/256, 256, 0, stream>>>(f1, fp1, 32, 88);
    repack_feat_kernel<<<(t2 + 255)/256, 256, 0, stream>>>(f2, fp2, 16, 44);
    repack_feat_kernel<<<(t3 + 255)/256, 256, 0, stream>>>(f3, fp3, 8, 22);
  }

  // geometry + gather
  project_softmax_kernel<<<(NPTS + 255)/256, 256, 0, stream>>>(spts, l2i, scw, ptsbuf, swbuf);
  sample_kernel<<<NPTS, 64, 0, stream>>>(fp0, fp1, fp2, fp3, ptsbuf, swbuf, x_bf);

  // params = query @ pg_w + pg_b   (1800 x 17408, K=256) -> bf16
  wmma_gemm_kernel<<<dim3((NPAR/16)/16, (BQ + 15)/16), 256, 0, stream>>>(
      q_bf, pgw_pk, pg_b, nullptr, nullptr, params, BQ, NPAR, 256);

  // per-(bq,g) mixing + layernorms
  xm_gemm_kernel<<<BQ*GG, 128, 0, stream>>>(x_bf, params, out1);
  ln_relu_xpose_kernel<<<BQ*GG, 256, 0, stream>>>(out1, y1t);
  sy_gemm_kernel<<<dim3(8, BQ*GG), 32, 0, stream>>>(params, y1t, out2);
  ln_relu_kernel<<<BQ*GG, 256, 0, stream>>>(out2, xf, 2048);

  // out = query + xf @ out_w + out_b   (1800 x 256, K=8192) -> fp32
  wmma_gemm_kernel<<<dim3((256/16)/16, (BQ + 15)/16), 256, 0, stream>>>(
      xf, outw_pk, out_b, query, out, nullptr, BQ, 256, 8192);

  (void)in_sizes; (void)n_in; (void)out_size; (void)ws_size;
}
